// MultiHeadAttention_18580028522901
// MI455X (gfx1250) — matrix operand order repaired and checked by reading, not compiled, not run on hardware
//
#include <hip/hip_runtime.h>
#include <hip/hip_bf16.h>

// MI455X / gfx1250, wave32. All heavy math via v_wmma_f32_16x16x32_bf16.
//
// Pipeline:
//  K1 qkv_proj : Q = X@Wq^T+bq (bf16, row-major), K likewise, V stored
//                transposed per attention group: Vt[g][d][t].
//  K2 attn     : per (group, 64-row q tile) flash attention, fp32 online
//                softmax, P@Vt via WMMA, output written as Yt[g][d][t]
//                (which IS the reference's swapaxes+reshape layout, flat
//                row-major (8192,512)).
//  K3 out_proj : out = Yt@Wo^T + bo, fp32 result.

typedef __attribute__((ext_vector_type(16))) __bf16 v16bf;
typedef __attribute__((ext_vector_type(8)))  __bf16 v8bf;
typedef __attribute__((ext_vector_type(8)))  float  v8f;

#define HID     512
#define NGRP    32
#define TPG     2048
#define M_TOTAL 8192  // B*S

static __device__ inline v8f zero8() {
  v8f z;
#pragma unroll
  for (int i = 0; i < 8; ++i) z[i] = 0.0f;
  return z;
}

static __device__ inline v8f wmma_bf16(v16bf a, v16bf b, v8f c) {
  // (neg_a, A, neg_b, B, c_mod, C, reuse_a, reuse_b)
  return __builtin_amdgcn_wmma_f32_16x16x32_bf16(false, a, false, b, (short)0, c,
                                                 false, false);
}

static __device__ inline v16bf cat8(v8bf lo, v8bf hi) {
  return __builtin_shufflevector(lo, hi, 0, 1, 2, 3, 4, 5, 6, 7,
                                 8, 9, 10, 11, 12, 13, 14, 15);
}

static __device__ inline v8bf cvt8(v8f x) {
  v8bf r;
#pragma unroll
  for (int i = 0; i < 8; ++i) r[i] = (__bf16)x[i];
  return r;
}

// ---------------------------------------------------------------------------
// Kernel 1: fused QKV projection.  C(8192x512) = X @ W^T + b.
// Block: 128 threads = 4 waves.  Block tile 128(M)x64(N); wave tile 32x64.
// gridDim.z selects Wq/Wk/Wv.  V output is stored transposed per group.
// ---------------------------------------------------------------------------
__global__ __launch_bounds__(128) void qkv_proj_kernel(
    const float* __restrict__ X,
    const float* __restrict__ Wq, const float* __restrict__ bq,
    const float* __restrict__ Wk, const float* __restrict__ bk,
    const float* __restrict__ Wv, const float* __restrict__ bv,
    __bf16* __restrict__ Qb, __bf16* __restrict__ Kb, __bf16* __restrict__ Vt)
{
  const int which = blockIdx.z;
  const float* Wm   = (which == 0) ? Wq : (which == 1) ? Wk : Wv;
  const float* bias = (which == 0) ? bq : (which == 1) ? bk : bv;

  const int lane  = threadIdx.x & 31;
  const int wv    = threadIdx.x >> 5;
  const int m0    = blockIdx.x * 128 + wv * 32;
  const int n0    = blockIdx.y * 64;
  const int l16   = lane & 15;
  const int hiA   = (lane >> 4) * 16;   // A-frag K-half select
  const int hiB   = (lane >> 4) * 16;   // B-frag K-half select
  const int rbase = (lane >> 4) * 8;    // C-frag row-half select

  v8f acc[2][4];
#pragma unroll
  for (int mt = 0; mt < 2; ++mt)
#pragma unroll
    for (int j = 0; j < 4; ++j) acc[mt][j] = zero8();

  for (int kk = 0; kk < HID; kk += 32) {
    v16bf afr[2];
#pragma unroll
    for (int mt = 0; mt < 2; ++mt) {
      const int row  = m0 + mt * 16 + l16;
      const float* p = X + (size_t)row * HID + kk + hiA;
      v8f lo = *(const v8f*)p;          // K = kk+hiA .. +7
      v8f hi = *(const v8f*)(p + 8);    // K = kk+hiA+8 .. +15
      afr[mt] = cat8(cvt8(lo), cvt8(hi));
    }
#pragma unroll
    for (int j = 0; j < 4; ++j) {
      // B[k,n] = W[n,k] -> lane n holds 16 contiguous k of W row n
      const int n    = n0 + j * 16 + l16;
      const float* p = Wm + (size_t)n * HID + kk + hiB;
      v8f x0 = *(const v8f*)p;
      v8f x1 = *(const v8f*)(p + 8);
      v16bf bfr = cat8(cvt8(x0), cvt8(x1));
#pragma unroll
      for (int mt = 0; mt < 2; ++mt)
        acc[mt][j] = wmma_bf16(afr[mt], bfr, acc[mt][j]);
    }
  }

  // epilogue: + bias, convert to bf16, store
#pragma unroll
  for (int j = 0; j < 4; ++j) {
    const int n     = n0 + j * 16 + l16;
    const float bns = bias[n];
#pragma unroll
    for (int mt = 0; mt < 2; ++mt) {
#pragma unroll
      for (int r = 0; r < 8; ++r) {
        const int   m = m0 + mt * 16 + rbase + r;
        const float v = acc[mt][j][r] + bns;
        if (which == 2) {
          // transposed per-group store: Vt[g][d][t]
          const int g = m >> 8;
          const int t = ((m & 255) << 3) | (n >> 6);
          const int d = n & 63;
          Vt[(size_t)g * 131072 + (size_t)d * 2048 + t] = (__bf16)v;
        } else {
          __bf16* O = (which == 0) ? Qb : Kb;
          O[(size_t)m * HID + n] = (__bf16)v;  // row-major == grouped (g,t,d)
        }
      }
    }
  }
}

// ---------------------------------------------------------------------------
// Kernel 2: flash attention per (group, 64-query tile). 4 waves, 16 q-rows
// each, full d=64 kept in registers. Online softmax in fp32. Output written
// transposed (d-major) through LDS -> coalesced b128 stores to Yt.
// ---------------------------------------------------------------------------
__global__ __launch_bounds__(128) void attn_kernel(
    const __bf16* __restrict__ Qb, const __bf16* __restrict__ Kb,
    const __bf16* __restrict__ Vt, const int* __restrict__ pm,
    __bf16* __restrict__ Yt)
{
  const int g    = blockIdx.y;
  const int t0   = blockIdx.x * 64;
  const int lane = threadIdx.x & 31;
  const int w    = threadIdx.x >> 5;
  const int l16  = lane & 15;
  const int hiA  = (lane >> 4) * 16;
  const int hiB  = (lane >> 4) * 16;
  const int rbase = (lane >> 4) * 8;

  const __bf16* Qg = Qb + (size_t)g * 131072;
  const __bf16* Kg = Kb + (size_t)g * 131072;
  const __bf16* Vg = Vt + (size_t)g * 131072;

  __shared__ alignas(16) __bf16 Plds[4][16][64];  // per-wave P staging (C->A layout)
  __shared__ alignas(16) __bf16 Olds[64][80];     // d-major output tile (+pad)

  // Q fragments: 16 rows x 64 d, two 32-wide K-chunks
  v16bf qf[2];
#pragma unroll
  for (int c = 0; c < 2; ++c) {
    const int t = t0 + w * 16 + l16;
    const __bf16* p = Qg + (size_t)t * 64 + c * 32 + hiA;
    qf[c] = cat8(*(const v8bf*)p, *(const v8bf*)(p + 8));
  }

  v8f   o[4];
  float mi[8], li[8];
#pragma unroll
  for (int j = 0; j < 4; ++j) o[j] = zero8();
#pragma unroll
  for (int r = 0; r < 8; ++r) { mi[r] = -__builtin_inff(); li[r] = 0.0f; }

  const float scale = 0.044194173824159216f;  // 1/sqrt(HIDDEN) per reference
  const float LOG2E = 1.4426950408889634f;

  for (int kb = 0; kb < TPG; kb += 64) {
    // ---- S = Q @ K^T : 16 x 64 (4 N-tiles) ----
    v8f s[4];
#pragma unroll
    for (int j = 0; j < 4; ++j) {
      v8f a = zero8();
      const int tk = kb + j * 16 + l16;  // lane n -> key row (contig. in d)
#pragma unroll
      for (int c = 0; c < 2; ++c) {
        v16bf bf = *(const v16bf*)(Kg + (size_t)tk * 64 + c * 32 + hiB);
        a = wmma_bf16(qf[c], bf, a);
      }
      s[j] = a;
    }

    // scale + padding mask (-inf where mask==0), per-lane row max
    float rowmax[8];
#pragma unroll
    for (int r = 0; r < 8; ++r) rowmax[r] = -__builtin_inff();
#pragma unroll
    for (int j = 0; j < 4; ++j) {
      const int  kcol = kb + j * 16 + l16;
      const bool ok   = (pm[kcol] != 0);
#pragma unroll
      for (int r = 0; r < 8; ++r) {
        float v   = ok ? s[j][r] * scale : -__builtin_inff();
        s[j][r]   = v;
        rowmax[r] = fmaxf(rowmax[r], v);
      }
    }
    // cross-lane reduce within each 16-lane half (rows split across halves)
#pragma unroll
    for (int r = 0; r < 8; ++r) {
      float v = rowmax[r];
      v = fmaxf(v, __shfl_xor(v, 1));
      v = fmaxf(v, __shfl_xor(v, 2));
      v = fmaxf(v, __shfl_xor(v, 4));
      v = fmaxf(v, __shfl_xor(v, 8));
      rowmax[r] = v;
    }

    float corr[8], lpart[8];
#pragma unroll
    for (int r = 0; r < 8; ++r) {
      const float mnew = fmaxf(mi[r], rowmax[r]);
      corr[r]  = exp2f((mi[r] - mnew) * LOG2E);
      mi[r]    = mnew;
      lpart[r] = 0.0f;
    }
#pragma unroll
    for (int j = 0; j < 4; ++j)
#pragma unroll
      for (int r = 0; r < 8; ++r) {
        const float p = exp2f((s[j][r] - mi[r]) * LOG2E);
        s[j][r] = p;
        lpart[r] += p;
      }
#pragma unroll
    for (int r = 0; r < 8; ++r) {
      float v = lpart[r];
      v += __shfl_xor(v, 1);
      v += __shfl_xor(v, 2);
      v += __shfl_xor(v, 4);
      v += __shfl_xor(v, 8);
      li[r] = li[r] * corr[r] + v;
    }
#pragma unroll
    for (int j = 0; j < 4; ++j)
#pragma unroll
      for (int r = 0; r < 8; ++r) o[j][r] *= corr[r];

    // P (C-layout) -> per-wave LDS -> reload in A-layout
#pragma unroll
    for (int j = 0; j < 4; ++j)
#pragma unroll
      for (int r = 0; r < 8; ++r)
        Plds[w][rbase + r][j * 16 + l16] = (__bf16)s[j][r];

    v16bf pf[2];
#pragma unroll
    for (int c = 0; c < 2; ++c) {
      const __bf16* p = &Plds[w][l16][c * 32 + hiA];
      pf[c] = cat8(*(const v8bf*)p, *(const v8bf*)(p + 8));
    }

    // ---- O += P @ V : B[k,d] = Vt[d][kb+k] (contiguous per lane) ----
#pragma unroll
    for (int j = 0; j < 4; ++j) {
      const int d = j * 16 + l16;
#pragma unroll
      for (int c = 0; c < 2; ++c) {
        v16bf bf = *(const v16bf*)(Vg + (size_t)d * 2048 + kb + c * 32 + hiB);
        o[j] = wmma_bf16(pf[c], bf, o[j]);
      }
    }
  }

  // normalize and stage transposed (d-major) output tile
#pragma unroll
  for (int j = 0; j < 4; ++j) {
    const int d = j * 16 + l16;
#pragma unroll
    for (int r = 0; r < 8; ++r) {
      const int tl = w * 16 + rbase + r;
      Olds[d][tl] = (__bf16)(o[j][r] / li[r]);
    }
  }
  __syncthreads();

  // coalesced b128 stores: Yt[g][d][t0..t0+63]
  for (int idx = threadIdx.x; idx < 64 * 8; idx += 128) {
    const int d  = idx >> 3;
    const int c8 = (idx & 7) << 3;
    v8bf row = *(const v8bf*)&Olds[d][c8];
    *(v8bf*)(Yt + (size_t)g * 131072 + (size_t)d * 2048 + t0 + c8) = row;
  }
}

// ---------------------------------------------------------------------------
// Kernel 3: output projection.  out(8192x512,f32) = Yt @ Wo^T + bo.
// ---------------------------------------------------------------------------
__global__ __launch_bounds__(128) void out_proj_kernel(
    const __bf16* __restrict__ Yt, const float* __restrict__ Wo,
    const float* __restrict__ bo, float* __restrict__ out)
{
  const int lane  = threadIdx.x & 31;
  const int wv    = threadIdx.x >> 5;
  const int m0    = blockIdx.x * 128 + wv * 32;
  const int n0    = blockIdx.y * 64;
  const int l16   = lane & 15;
  const int hiA   = (lane >> 4) * 16;
  const int hiB   = (lane >> 4) * 16;
  const int rbase = (lane >> 4) * 8;

  v8f acc[2][4];
#pragma unroll
  for (int mt = 0; mt < 2; ++mt)
#pragma unroll
    for (int j = 0; j < 4; ++j) acc[mt][j] = zero8();

  for (int kk = 0; kk < HID; kk += 32) {
    v16bf afr[2];
#pragma unroll
    for (int mt = 0; mt < 2; ++mt) {
      const int row = m0 + mt * 16 + l16;
      const __bf16* p = Yt + (size_t)row * HID + kk + hiA;
      afr[mt] = cat8(*(const v8bf*)p, *(const v8bf*)(p + 8));
    }
#pragma unroll
    for (int j = 0; j < 4; ++j) {
      const int n    = n0 + j * 16 + l16;
      const float* p = Wo + (size_t)n * HID + kk + hiB;
      v16bf bfr = cat8(cvt8(*(const v8f*)p), cvt8(*(const v8f*)(p + 8)));
#pragma unroll
      for (int mt = 0; mt < 2; ++mt)
        acc[mt][j] = wmma_bf16(afr[mt], bfr, acc[mt][j]);
    }
  }

#pragma unroll
  for (int j = 0; j < 4; ++j) {
    const int   n   = n0 + j * 16 + l16;
    const float bns = bo[n];
#pragma unroll
    for (int mt = 0; mt < 2; ++mt)
#pragma unroll
      for (int r = 0; r < 8; ++r) {
        const int m = m0 + mt * 16 + rbase + r;
        out[(size_t)m * HID + n] = acc[mt][j][r] + bns;
      }
  }
}

// ---------------------------------------------------------------------------
extern "C" void kernel_launch(void* const* d_in, const int* in_sizes, int n_in,
                              void* d_out, int out_size, void* d_ws, size_t ws_size,
                              hipStream_t stream) {
  (void)in_sizes; (void)n_in; (void)out_size; (void)ws_size;
  const float* X  = (const float*)d_in[0];
  const int*   pm = (const int*)d_in[1];
  const float* Wq = (const float*)d_in[2];
  const float* bq = (const float*)d_in[3];
  const float* Wk = (const float*)d_in[4];
  const float* bk = (const float*)d_in[5];
  const float* Wv = (const float*)d_in[6];
  const float* bv = (const float*)d_in[7];
  const float* Wo = (const float*)d_in[8];
  const float* bo = (const float*)d_in[9];
  float* out = (float*)d_out;

  // workspace: 4 x (8192*512) bf16 = 32 MB total
  __bf16* Qb = (__bf16*)d_ws;
  __bf16* Kb = Qb + (size_t)M_TOTAL * HID;
  __bf16* Vt = Kb + (size_t)M_TOTAL * HID;
  __bf16* Yt = Vt + (size_t)M_TOTAL * HID;

  dim3 blk(128);
  dim3 g1(M_TOTAL / 128, HID / 64, 3);
  qkv_proj_kernel<<<g1, blk, 0, stream>>>(X, Wq, bq, Wk, bk, Wv, bv, Qb, Kb, Vt);

  dim3 g2(TPG / 64, NGRP);
  attn_kernel<<<g2, blk, 0, stream>>>(Qb, Kb, Vt, pm, Yt);

  dim3 g3(M_TOTAL / 128, HID / 64);
  out_proj_kernel<<<g3, blk, 0, stream>>>(Yt, Wo, bo, out);
}
